// DiTeBlock_19911468384608
// MI455X (gfx1250) — compile-verified, matrix operand order checked by reading, not run on hardware
//
#include <hip/hip_runtime.h>
#include <math.h>

// Problem constants (match reference)
#define N_NODES 25000
#define N_EDGES 250000
#define HDIM    256
#define HEDIM   128
#define NHEADS  8
#define DHEAD   32
#define DISTD   128
#define FFH     1024
#define FFHE    512

typedef __attribute__((ext_vector_type(16))) __bf16 bf16x16;
typedef __attribute__((ext_vector_type(8)))  float  f32x8;

static __device__ __forceinline__ float silu_f(float v) { return v / (1.f + __expf(-v)); }

// pack two f32 -> one dword {bf16(hi)[31:16], bf16(lo)[15:0]} via v_cvt_pk_bf16_f32
static __device__ __forceinline__ unsigned pk_bf16(float lo, float hi) {
#if __has_builtin(__builtin_amdgcn_cvt_pk_bf16_f32)
    auto p = __builtin_amdgcn_cvt_pk_bf16_f32(lo, hi);
    return __builtin_bit_cast(unsigned, p);
#else
    union { __bf16 h[2]; unsigned u; } t;
    t.h[0] = (__bf16)lo; t.h[1] = (__bf16)hi;
    return t.u;
#endif
}

// ---------------------------------------------------------------------------
// Tiled GEMM: C[M,N] = act(A[M,K] @ B[K,N] + bias), f32 in/out, bf16 WMMA.
// Block = 256 threads = 8 waves (2M x 4N), tile = 32(M) x 128(N), K-step 32.
// Double-buffered LDS (ping-pong, K unrolled by 64 -> ONE barrier per K-step);
// next tile's global loads issue in the shadow of current ds_load + WMMA.
// OOB rows/cols are CLAMPED, not predicated: garbage only lands in
// accumulator lanes whose stores are masked in the epilogue, so EXEC stays
// all-ones through the WMMA loop (ISA requirement), no per-load branching.
// Requires K % 64 == 0 (true for every GEMM in this model).
// ACT: 0 = none, 1 = SiLU
// ---------------------------------------------------------------------------
template<int ACT>
__launch_bounds__(256)
__global__ void gemm_wmma_bf16(const float* __restrict__ A, const float* __restrict__ B,
                               const float* __restrict__ bias, float* __restrict__ C,
                               int M, int N, int K)
{
    __shared__ unsigned As32[2][32][19];    // [buf][m][k/2]  (19: coprime w/ 64 banks)
    __shared__ unsigned Bs32[2][128][19];   // [buf][n][k/2]

    const int tid   = threadIdx.x;
    const int lane  = tid & 31;
    const int wave  = tid >> 5;
    const int tileM = blockIdx.y << 5;
    const int tileN = blockIdx.x << 7;
    const int mbase = (wave >> 2) << 4;          // 0 / 16
    const int nbase = (wave & 3) << 5;           // 0 / 32 / 64 / 96
    const int mfrag = lane & 15;
    const int khalf = lane >> 4;

    // ---- fixed per-thread staging maps (addresses hoisted out of the loop) ----
    // A: row = tid/8, 8 threads x 2 float2 per row of 32 elems
    const int    arow = tid >> 3;
    const size_t agm  = (size_t)min(tileM + arow, M - 1);
    const float* Ap   = A + agm * (size_t)K + ((tid & 7) << 1);
    // B: n fixed per thread, k-pair base jp0 = tid/128 (0/1), rows kk+2*jp0+4*rep
    const int    bn   = tid & 127;
    const int    bjp0 = tid >> 7;
    const float* Bp   = B + (size_t)min(tileN + bn, N - 1) + (size_t)(bjp0 << 1) * N;

    f32x8 acc0 = {}, acc1 = {};

    auto stage = [&](int buf, int kk) {
        #pragma unroll
        for (int rep = 0; rep < 2; ++rep) {
            const float2 v = *(const float2*)&Ap[kk + (rep << 4)];
            As32[buf][arow][(tid & 7) + (rep << 3)] = pk_bf16(v.x, v.y);
        }
        #pragma unroll
        for (int rep = 0; rep < 8; ++rep) {
            size_t o = (size_t)(kk + (rep << 2)) * N;   // uniform per rep
            float lo = Bp[o];
            float hi = Bp[o + N];
            Bs32[buf][bn][bjp0 + (rep << 1)] = pk_bf16(lo, hi);
        }
    };

    union U { bf16x16 v; unsigned u[8]; };
    auto mm = [&](int buf) {
        U a, b0, b1;
        #pragma unroll
        for (int vg = 0; vg < 8; ++vg) {
            // v16bf dword vg holds K = {k, k+1}, k = (vg%4)*2 + khalf*8 + (vg/4)*16
            int k2 = (vg & 3) + (khalf << 2) + ((vg >> 2) << 3);
            a.u[vg]  = As32[buf][mbase + mfrag][k2];
            b0.u[vg] = Bs32[buf][nbase + mfrag][k2];
            b1.u[vg] = Bs32[buf][nbase + 16 + mfrag][k2];
        }
        acc0 = __builtin_amdgcn_wmma_f32_16x16x32_bf16(false, a.v, false, b0.v, (short)0, acc0, false, false);
        acc1 = __builtin_amdgcn_wmma_f32_16x16x32_bf16(false, a.v, false, b1.v, (short)0, acc1, false, false);
    };

    stage(0, 0);
    __syncthreads();
    for (int kk = 0; kk < K; kk += 64) {
        stage(1, kk + 32);                  // K%64==0 -> kk+32 < K always
        if (kk + 64 < K) {                  // uniform: prefetch the tile after next
            __builtin_prefetch(&Ap[kk + 64], 0, 1);
            __builtin_prefetch(&Bp[(size_t)(kk + 64) * N], 0, 1);
        }
        mm(0);
        __syncthreads();
        if (kk + 64 < K) stage(0, kk + 64); // uniform branch
        mm(1);
        __syncthreads();
    }

    // ---- epilogue: C/D layout VGPR r -> M = r + 8*khalf ----
    const int ncol0 = tileN + nbase + mfrag;
    const int ncol1 = ncol0 + 16;
    #pragma unroll
    for (int r = 0; r < 8; ++r) {
        int gm = tileM + mbase + r + (khalf << 3);
        if (gm < M) {
            if (ncol0 < N) {
                float v = acc0[r];
                if (bias) v += bias[ncol0];
                if (ACT == 1) v = silu_f(v);
                C[(size_t)gm * N + ncol0] = v;
            }
            if (ncol1 < N) {
                float v = acc1[r];
                if (bias) v += bias[ncol1];
                if (ACT == 1) v = silu_f(v);
                C[(size_t)gm * N + ncol1] = v;
            }
        }
    }
}

// ---------------------------------------------------------------------------
// LayerNorm + adaLN modulation: out = ln(x)*(1+ada[:,scOff:])+ada[:,shOff:]
// one wave32 per row
// ---------------------------------------------------------------------------
__launch_bounds__(256)
__global__ void k_ln_mod(const float* __restrict__ x, const float* __restrict__ ada,
                         int adaStride, int shOff, int scOff,
                         float* __restrict__ out, int rows, int C)
{
    int lane = threadIdx.x & 31;
    int row  = blockIdx.x * 8 + (threadIdx.x >> 5);
    if (row >= rows) return;
    const float* xr = x + (size_t)row * C;
    float s = 0.f, ss = 0.f;
    for (int c = lane; c < C; c += 32) { float v = xr[c]; s += v; ss += v * v; }
    #pragma unroll
    for (int o = 16; o > 0; o >>= 1) { s += __shfl_xor(s, o, 32); ss += __shfl_xor(ss, o, 32); }
    float mean = s / C;
    float var  = ss / C - mean * mean;
    float rstd = rsqrtf(var + 1e-6f);
    const float* ar = ada + (size_t)row * adaStride;
    float* orow = out + (size_t)row * C;
    for (int c = lane; c < C; c += 32) {
        float v = (xr[c] - mean) * rstd;
        orow[c] = v * (1.f + ar[scOff + c]) + ar[shOff + c];
    }
}

// ---------------------------------------------------------------------------
// Elementwise / gather / scatter kernels
// ---------------------------------------------------------------------------
__global__ void k_silu(const float* __restrict__ in, float* __restrict__ out, size_t n) {
    size_t i = (size_t)blockIdx.x * blockDim.x + threadIdx.x;
    if (i < n) out[i] = silu_f(in[i]);
}

__global__ void k_fill(float* __restrict__ p, float v, size_t n) {
    size_t i = (size_t)blockIdx.x * blockDim.x + threadIdx.x;
    if (i < n) p[i] = v;
}

// msg_in = [x_norm[src] | x_norm[tgt] | e_norm | dist]   (E x 768)
__global__ void k_concat_msg(const float* __restrict__ xn, const float* __restrict__ en,
                             const float* __restrict__ dist, const int* __restrict__ ei,
                             float* __restrict__ out) {
    size_t i = (size_t)blockIdx.x * blockDim.x + threadIdx.x;
    if (i >= (size_t)N_EDGES * 768) return;
    int e = (int)(i / 768), c = (int)(i % 768);
    float v;
    if (c < 256)       v = xn[(size_t)ei[e] * 256 + c];
    else if (c < 512)  v = xn[(size_t)ei[N_EDGES + e] * 256 + (c - 256)];
    else if (c < 640)  v = en[(size_t)e * 128 + (c - 512)];
    else               v = dist[(size_t)e * 128 + (c - 640)];
    out[i] = v;
}

// scatter-mean numerator + counts over src
__global__ void k_scatter_msg(const float* __restrict__ msg, const int* __restrict__ ei,
                              float* __restrict__ xagg, float* __restrict__ cnt) {
    size_t i = (size_t)blockIdx.x * blockDim.x + threadIdx.x;
    if (i >= (size_t)N_EDGES * 256) return;
    int e = (int)(i >> 8), c = (int)(i & 255);
    int src = ei[e];
    atomicAdd(&xagg[(size_t)src * 256 + c], msg[i]);
    if (c == 0) atomicAdd(&cnt[src], 1.f);
}

__global__ void k_div_cnt(float* __restrict__ xagg, const float* __restrict__ cnt) {
    size_t i = (size_t)blockIdx.x * blockDim.x + threadIdx.x;
    if (i >= (size_t)N_NODES * 256) return;
    xagg[i] /= fmaxf(cnt[i >> 8], 1.f);
}

// alpha = (Q[src].K[tgt] + DH*bias)/sqrt(DH)     (E x NH)
__global__ void k_attn_logits(const float* __restrict__ qkv, const float* __restrict__ pb,
                              const int* __restrict__ ei, float* __restrict__ alpha) {
    size_t i = (size_t)blockIdx.x * blockDim.x + threadIdx.x;
    if (i >= (size_t)N_EDGES * NHEADS) return;
    int e = (int)(i >> 3), h = (int)(i & 7);
    const float* q = qkv + (size_t)ei[e] * 768 + h * 32;
    const float* k = qkv + (size_t)ei[N_EDGES + e] * 768 + 256 + h * 32;
    float dot = 0.f;
    #pragma unroll
    for (int d = 0; d < 32; ++d) dot += q[d] * k[d];
    alpha[i] = (dot + 32.f * pb[i]) * 0.17677669529663687f; // 1/sqrt(32)
}

static __device__ __forceinline__ void atomicMaxFloat(float* addr, float val) {
    int* ia = (int*)addr;
    int old = __float_as_int(*addr);
    while (__int_as_float(old) < val) {
        int assumed = old;
        old = atomicCAS(ia, assumed, __float_as_int(val));
        if (old == assumed) break;
    }
}

__global__ void k_seg_max(const float* __restrict__ alpha, const int* __restrict__ ei,
                          float* __restrict__ mx) {
    size_t i = (size_t)blockIdx.x * blockDim.x + threadIdx.x;
    if (i >= (size_t)N_EDGES * NHEADS) return;
    int e = (int)(i >> 3), h = (int)(i & 7);
    atomicMaxFloat(&mx[(size_t)ei[N_EDGES + e] * NHEADS + h], alpha[i]);
}

__global__ void k_seg_expsum(float* __restrict__ alpha, const int* __restrict__ ei,
                             const float* __restrict__ mx, float* __restrict__ sum) {
    size_t i = (size_t)blockIdx.x * blockDim.x + threadIdx.x;
    if (i >= (size_t)N_EDGES * NHEADS) return;
    int e = (int)(i >> 3), h = (int)(i & 7);
    int tgt = ei[N_EDGES + e];
    float ex = __expf(alpha[i] - mx[(size_t)tgt * NHEADS + h]);
    alpha[i] = ex;
    atomicAdd(&sum[(size_t)tgt * NHEADS + h], ex);
}

// out[tgt] += V[tgt] * alpha/sum     (accumulate into N x 256)
__global__ void k_attn_scatter(const float* __restrict__ qkv, const float* __restrict__ alpha,
                               const float* __restrict__ sum, const int* __restrict__ ei,
                               float* __restrict__ out) {
    size_t i = (size_t)blockIdx.x * blockDim.x + threadIdx.x;
    if (i >= (size_t)N_EDGES * 256) return;
    int e = (int)(i >> 8), rem = (int)(i & 255);     // rem = h*32 + d
    int h = rem >> 5;
    int tgt = ei[N_EDGES + e];
    float w = alpha[(size_t)e * NHEADS + h] / (sum[(size_t)tgt * NHEADS + h] + 1e-16f);
    atomicAdd(&out[(size_t)tgt * 256 + rem], qkv[(size_t)tgt * 768 + 512 + rem] * w);
}

// out = base + ada[:, gOff:gOff+C] * y
__global__ void k_resid(const float* __restrict__ base, const float* __restrict__ ada,
                        int adaStride, int gOff, const float* __restrict__ y,
                        float* __restrict__ out, size_t rows, int C) {
    size_t i = (size_t)blockIdx.x * blockDim.x + threadIdx.x;
    if (i >= rows * (size_t)C) return;
    size_t r = i / C; int c = (int)(i % C);
    out[i] = base[i] + ada[r * adaStride + gOff + c] * y[i];
}

// ysum = y[src] + y[tgt]    (E x 256)
__global__ void k_edge_ysum(const float* __restrict__ y, const int* __restrict__ ei,
                            float* __restrict__ out) {
    size_t i = (size_t)blockIdx.x * blockDim.x + threadIdx.x;
    if (i >= (size_t)N_EDGES * 256) return;
    int e = (int)(i >> 8), c = (int)(i & 255);
    out[i] = y[(size_t)ei[e] * 256 + c] + y[(size_t)ei[N_EDGES + e] * 256 + c];
}

// cat_e = [edge_mid | dist]   (E x 256)
__global__ void k_concat_edge(const float* __restrict__ em, const float* __restrict__ dist,
                              float* __restrict__ out) {
    size_t i = (size_t)blockIdx.x * blockDim.x + threadIdx.x;
    if (i >= (size_t)N_EDGES * 256) return;
    int e = (int)(i >> 8), c = (int)(i & 255);
    out[i] = (c < 128) ? em[(size_t)e * 128 + c] : dist[(size_t)e * 128 + (c - 128)];
}

// swiglu: out[r,c] = silu(h[r,c]) * h[r,F+c]
__global__ void k_swiglu(const float* __restrict__ h, float* __restrict__ out,
                         size_t rows, int F) {
    size_t i = (size_t)blockIdx.x * blockDim.x + threadIdx.x;
    if (i >= rows * (size_t)F) return;
    size_t r = i / F; int c = (int)(i % F);
    const float* hr = h + r * (size_t)(2 * F);
    out[i] = silu_f(hr[c]) * hr[F + c];
}

// ---------------------------------------------------------------------------
// Host orchestration
// ---------------------------------------------------------------------------
static inline dim3 g1(size_t n) { return dim3((unsigned)((n + 255) / 256)); }
static inline dim3 gemm_grid(int M, int N) { return dim3((N + 127) / 128, (M + 31) / 32); }

extern "C" void kernel_launch(void* const* d_in, const int* in_sizes, int n_in,
                              void* d_out, int out_size, void* d_ws, size_t ws_size,
                              hipStream_t stream) {
    (void)in_sizes; (void)n_in; (void)out_size; (void)ws_size;
    const float* x        = (const float*)d_in[1];
    const float* t_emb_h  = (const float*)d_in[2];
    const float* edge_attr= (const float*)d_in[3];
    const int*   ei       = (const int*)  d_in[4];
    const float* t_emb_e  = (const float*)d_in[5];
    const float* dist     = (const float*)d_in[6];
    const float* W_ada  = (const float*)d_in[7];  const float* b_ada  = (const float*)d_in[8];
    const float* W_adae = (const float*)d_in[9];  const float* b_adae = (const float*)d_in[10];
    const float* W_fe0  = (const float*)d_in[11]; const float* b_fe0  = (const float*)d_in[12];
    const float* W_fe1  = (const float*)d_in[13]; const float* b_fe1  = (const float*)d_in[14];
    const float* W_qkv  = (const float*)d_in[15];
    const float* W_out  = (const float*)d_in[16];
    const float* W_pb   = (const float*)d_in[17];
    const float* W_e0   = (const float*)d_in[18];
    const float* W_e1   = (const float*)d_in[19];
    const float* W_ffn1 = (const float*)d_in[20]; const float* W_ffn2 = (const float*)d_in[21];
    const float* W_effn1= (const float*)d_in[22]; const float* W_effn2= (const float*)d_in[23];

    // Workspace bump allocator (floats), with arena reuse for the giants
    size_t off = 0;
    float* base = (float*)d_ws;
    auto alloc = [&](size_t n) { float* p = base + off; off += n; return p; };
    float* big1 = alloc((size_t)N_EDGES * 1024);  // msg_in (E*768) -> effn1 (E*1024)
    float* big2 = alloc((size_t)N_EDGES * 512);   // hid -> ysum -> cat_e -> eswig
    float* big3 = alloc((size_t)N_EDGES * 256);   // messages -> ffn1 (N*2048)
    float* adaH   = alloc((size_t)N_NODES * 6 * HDIM);
    float* adaE   = alloc((size_t)N_EDGES * 6 * HEDIM);
    float* sTeH   = alloc((size_t)N_NODES * HDIM);
    float* sTeE   = alloc((size_t)N_EDGES * HEDIM);
    float* sDist  = alloc((size_t)N_EDGES * DISTD);
    float* xnorm  = alloc((size_t)N_NODES * HDIM);
    float* enorm  = alloc((size_t)N_EDGES * HEDIM);
    float* xagg   = alloc((size_t)N_NODES * HDIM);
    float* cnt    = alloc((size_t)N_NODES);
    float* qkv    = alloc((size_t)N_NODES * 3 * HDIM);
    float* pbias  = alloc((size_t)N_EDGES * NHEADS);
    float* alpha  = alloc((size_t)N_EDGES * NHEADS);
    float* segmax = alloc((size_t)N_NODES * NHEADS);
    float* segsum = alloc((size_t)N_NODES * NHEADS);
    float* attnout= alloc((size_t)N_NODES * HDIM);
    float* yb     = alloc((size_t)N_NODES * HDIM);
    float* xmid   = alloc((size_t)N_NODES * HDIM);
    float* e0out  = alloc((size_t)N_EDGES * HEDIM);
    float* edgemid= alloc((size_t)N_EDGES * HEDIM);
    float* xn     = alloc((size_t)N_NODES * HDIM);
    float* swig   = alloc((size_t)N_NODES * FFH);
    float* ffn2o  = alloc((size_t)N_NODES * HDIM);
    float* e_in   = alloc((size_t)N_EDGES * HEDIM);
    float* en     = alloc((size_t)N_EDGES * HEDIM);
    float* effn2o = alloc((size_t)N_EDGES * HEDIM);

    float* xfinal = (float*)d_out;                                 // N x 256
    float* efinal = (float*)d_out + (size_t)N_NODES * HDIM;        // E x 128

    float* msgin = big1;  float* effn1 = big1;
    float* hid   = big2;  float* ysum  = big2; float* cat_e = big2; float* eswig = big2;
    float* msgs  = big3;  float* ffn1  = big3;

    // --- adaLN modulation ---
    k_silu<<<g1((size_t)N_NODES*HDIM), 256, 0, stream>>>(t_emb_h, sTeH, (size_t)N_NODES*HDIM);
    k_silu<<<g1((size_t)N_EDGES*HEDIM), 256, 0, stream>>>(t_emb_e, sTeE, (size_t)N_EDGES*HEDIM);
    k_silu<<<g1((size_t)N_EDGES*DISTD), 256, 0, stream>>>(dist, sDist, (size_t)N_EDGES*DISTD);
    gemm_wmma_bf16<0><<<gemm_grid(N_NODES, 6*HDIM), 256, 0, stream>>>(sTeH, W_ada, b_ada, adaH, N_NODES, 6*HDIM, HDIM);
    gemm_wmma_bf16<0><<<gemm_grid(N_EDGES, 6*HEDIM), 256, 0, stream>>>(sTeE, W_adae, b_adae, adaE, N_EDGES, 6*HEDIM, HEDIM);

    // --- modulated LayerNorms (msa) ---
    k_ln_mod<<<dim3((N_NODES+7)/8), 256, 0, stream>>>(x, adaH, 6*HDIM, 0, HDIM, xnorm, N_NODES, HDIM);
    k_ln_mod<<<dim3((N_EDGES+7)/8), 256, 0, stream>>>(edge_attr, adaE, 6*HEDIM, 0, HEDIM, enorm, N_EDGES, HEDIM);

    // --- message MLP ---
    k_concat_msg<<<g1((size_t)N_EDGES*768), 256, 0, stream>>>(xnorm, enorm, dist, ei, msgin);
    gemm_wmma_bf16<1><<<gemm_grid(N_EDGES, HDIM), 256, 0, stream>>>(msgin, W_fe0, b_fe0, hid, N_EDGES, HDIM, 768);
    gemm_wmma_bf16<0><<<gemm_grid(N_EDGES, HDIM), 256, 0, stream>>>(hid, W_fe1, b_fe1, msgs, N_EDGES, HDIM, HDIM);

    // --- scatter mean over src ---
    k_fill<<<g1((size_t)N_NODES*HDIM), 256, 0, stream>>>(xagg, 0.f, (size_t)N_NODES*HDIM);
    k_fill<<<g1((size_t)N_NODES), 256, 0, stream>>>(cnt, 0.f, (size_t)N_NODES);
    k_scatter_msg<<<g1((size_t)N_EDGES*256), 256, 0, stream>>>(msgs, ei, xagg, cnt);
    k_div_cnt<<<g1((size_t)N_NODES*256), 256, 0, stream>>>(xagg, cnt);

    // --- attention ---
    gemm_wmma_bf16<0><<<gemm_grid(N_NODES, 3*HDIM), 256, 0, stream>>>(xagg, W_qkv, nullptr, qkv, N_NODES, 3*HDIM, HDIM);
    gemm_wmma_bf16<0><<<gemm_grid(N_EDGES, NHEADS), 256, 0, stream>>>(sDist, W_pb, nullptr, pbias, N_EDGES, NHEADS, DISTD);
    k_attn_logits<<<g1((size_t)N_EDGES*NHEADS), 256, 0, stream>>>(qkv, pbias, ei, alpha);
    k_fill<<<g1((size_t)N_NODES*NHEADS), 256, 0, stream>>>(segmax, -1e30f, (size_t)N_NODES*NHEADS);
    k_fill<<<g1((size_t)N_NODES*NHEADS), 256, 0, stream>>>(segsum, 0.f, (size_t)N_NODES*NHEADS);
    k_seg_max<<<g1((size_t)N_EDGES*NHEADS), 256, 0, stream>>>(alpha, ei, segmax);
    k_seg_expsum<<<g1((size_t)N_EDGES*NHEADS), 256, 0, stream>>>(alpha, ei, segmax, segsum);
    k_fill<<<g1((size_t)N_NODES*HDIM), 256, 0, stream>>>(attnout, 0.f, (size_t)N_NODES*HDIM);
    k_attn_scatter<<<g1((size_t)N_EDGES*256), 256, 0, stream>>>(qkv, alpha, segsum, ei, attnout);
    gemm_wmma_bf16<0><<<gemm_grid(N_NODES, HDIM), 256, 0, stream>>>(attnout, W_out, nullptr, yb, N_NODES, HDIM, HDIM);

    // --- residuals (msa) ---
    k_resid<<<g1((size_t)N_NODES*HDIM), 256, 0, stream>>>(x, adaH, 6*HDIM, 2*HDIM, yb, xmid, N_NODES, HDIM);
    k_edge_ysum<<<g1((size_t)N_EDGES*256), 256, 0, stream>>>(yb, ei, ysum);
    gemm_wmma_bf16<0><<<gemm_grid(N_EDGES, HEDIM), 256, 0, stream>>>(ysum, W_e0, nullptr, e0out, N_EDGES, HEDIM, HDIM);
    k_resid<<<g1((size_t)N_EDGES*HEDIM), 256, 0, stream>>>(edge_attr, adaE, 6*HEDIM, 2*HEDIM, e0out, edgemid, N_EDGES, HEDIM);

    // --- node SwiGLU FFN ---
    k_ln_mod<<<dim3((N_NODES+7)/8), 256, 0, stream>>>(xmid, adaH, 6*HDIM, 3*HDIM, 4*HDIM, xn, N_NODES, HDIM);
    gemm_wmma_bf16<0><<<gemm_grid(N_NODES, 2*FFH), 256, 0, stream>>>(xn, W_ffn1, nullptr, ffn1, N_NODES, 2*FFH, HDIM);
    k_swiglu<<<g1((size_t)N_NODES*FFH), 256, 0, stream>>>(ffn1, swig, N_NODES, FFH);
    gemm_wmma_bf16<0><<<gemm_grid(N_NODES, HDIM), 256, 0, stream>>>(swig, W_ffn2, nullptr, ffn2o, N_NODES, HDIM, FFH);
    k_resid<<<g1((size_t)N_NODES*HDIM), 256, 0, stream>>>(xmid, adaH, 6*HDIM, 5*HDIM, ffn2o, xfinal, N_NODES, HDIM);

    // --- edge SwiGLU FFN ---
    k_concat_edge<<<g1((size_t)N_EDGES*256), 256, 0, stream>>>(edgemid, dist, cat_e);
    gemm_wmma_bf16<0><<<gemm_grid(N_EDGES, HEDIM), 256, 0, stream>>>(cat_e, W_e1, nullptr, e_in, N_EDGES, HEDIM, HEDIM+DISTD);
    k_ln_mod<<<dim3((N_EDGES+7)/8), 256, 0, stream>>>(e_in, adaE, 6*HEDIM, 3*HEDIM, 4*HEDIM, en, N_EDGES, HEDIM);
    gemm_wmma_bf16<0><<<gemm_grid(N_EDGES, 2*FFHE), 256, 0, stream>>>(en, W_effn1, nullptr, effn1, N_EDGES, 2*FFHE, HEDIM);
    k_swiglu<<<g1((size_t)N_EDGES*FFHE), 256, 0, stream>>>(effn1, eswig, N_EDGES, FFHE);
    gemm_wmma_bf16<0><<<gemm_grid(N_EDGES, HEDIM), 256, 0, stream>>>(eswig, W_effn2, nullptr, effn2o, N_EDGES, HEDIM, FFHE);
    k_resid<<<g1((size_t)N_EDGES*HEDIM), 256, 0, stream>>>(edgemid, adaE, 6*HEDIM, 5*HEDIM, effn2o, efinal, N_EDGES, HEDIM);
}
